// BertHAttention1D_59227599012024
// MI455X (gfx1250) — matrix operand order repaired and checked by reading, not compiled, not run on hardware
//
#include <hip/hip_runtime.h>
#include <cstdint>
#include <cstddef>

// ---------------------------------------------------------------------------
// BERT H-Attention for MI455X (gfx1250, wave32, WMMA 16x16x32 f16)
//
// Pipeline:
//   1) cvt_f32_f16    : hidden_states f32 -> f16
//   2) transpose_w    : W f32 [k][n] -> Wt f16 [n][k]   (once per matrix)
//   3) qkv_gemm       : X @ W + b via WMMA f16; tiles staged with
//                       GLOBAL_LOAD_ASYNC_TO_LDS_B128 (ASYNCcnt pipeline);
//                       epilogue writes head-split f16 [64][4096][64]
//   4) coarsen        : per-level mean-pool q,k / sum-pool v      (x6)
//   5) hatt_level     : per 16-token block: S = q k^T (2 WMMA),
//                       rowmax/exp/rowsum via shfl_xor butterflies,
//                       y = exp(S) v (4 WMMA, K padded 16->32),
//                       fused upsample-accumulate from coarser level (x7)
//   6) finalize       : out = Y / (A + eps), head merge
// ---------------------------------------------------------------------------

#define BB   4
#define NN   4096
#define HH   16
#define DHD  64
#define DIMD 1024
#define MROWS (BB*NN)   // 16384
#define BHD   (BB*HH)   // 64

typedef _Float16 h2  __attribute__((ext_vector_type(2)));
typedef _Float16 h8  __attribute__((ext_vector_type(8)));
typedef _Float16 h16 __attribute__((ext_vector_type(16)));
typedef float    f8  __attribute__((ext_vector_type(8)));
typedef float    f4  __attribute__((ext_vector_type(4)));

__device__ __forceinline__ h16 hcat(h8 a, h8 b) {
  h16 r;
#pragma unroll
  for (int i = 0; i < 8; i++) { r[i] = a[i]; r[i + 8] = b[i]; }
  return r;
}

__device__ __forceinline__ f8 f8zero() {
  f8 z;
#pragma unroll
  for (int i = 0; i < 8; i++) z[i] = 0.0f;
  return z;
}

__device__ __forceinline__ f8 wmma_f16(h16 a, h16 b, f8 c) {
  // v_wmma_f32_16x16x32_f16  (neg_a, A, neg_b, B, c_mod, C, reuse_a, reuse_b)
  return __builtin_amdgcn_wmma_f32_16x16x32_f16(false, a, false, b, (short)0, c,
                                                false, false);
}

// Async global->LDS copy, 16B per lane. VDST operand carries the LDS byte
// address; per the CDNA5 aperture rule (ISA 10.2) the low 32 bits of a generic
// __shared__ pointer ARE the LDS byte address.
__device__ __forceinline__ void async_b128(void* lds, const void* gsrc) {
  uint32_t l = (uint32_t)(uintptr_t)lds;
  asm volatile("global_load_async_to_lds_b128 %0, %1, off"
               :
               : "v"(l), "v"(gsrc)
               : "memory");
}

// ---------------------------------------------------------------------------
// 1) f32 -> f16 conversion (8 elements / thread)
// ---------------------------------------------------------------------------
__global__ void cvt_f32_f16(const float* __restrict__ in,
                            _Float16* __restrict__ out, size_t n8) {
  size_t i = (size_t)blockIdx.x * blockDim.x + threadIdx.x;
  if (i >= n8) return;
  f4 a = *(const f4*)(in + i * 8);
  f4 b = *(const f4*)(in + i * 8 + 4);
  h8 o;
#pragma unroll
  for (int e = 0; e < 4; e++) { o[e] = (_Float16)a[e]; o[e + 4] = (_Float16)b[e]; }
  *(h8*)(out + i * 8) = o;
}

// ---------------------------------------------------------------------------
// 2) W f32 [k][n] -> Wt f16 [n][k], 64x64 LDS tile transpose
// ---------------------------------------------------------------------------
__global__ __launch_bounds__(256) void transpose_w(
    const float* __restrict__ W, _Float16* __restrict__ Wt) {
  __shared__ float tl[64][65];
  const int t = threadIdx.x;
  const int n0 = blockIdx.x * 64;
  const int k0 = blockIdx.y * 64;
#pragma unroll
  for (int it = 0; it < 16; it++) {
    int slot = t + it * 256;           // 4096 elements
    int kr = slot >> 6, nc = slot & 63;
    tl[kr][nc] = W[(size_t)(k0 + kr) * DIMD + n0 + nc];
  }
  __syncthreads();
#pragma unroll
  for (int it = 0; it < 8; it++) {
    int slot = t + it * 256;           // 2048 pairs
    int nr = slot >> 5, kc = (slot & 31) * 2;
    h2 p;
    p[0] = (_Float16)tl[kc][nr];
    p[1] = (_Float16)tl[kc + 1][nr];
    *(h2*)&Wt[(size_t)(n0 + nr) * DIMD + k0 + kc] = p;
  }
}

// ---------------------------------------------------------------------------
// 3) WMMA GEMM: Out[bh][tok][d] = f16( (X @ W + b) * scale ), head-split
//    WG = 256 threads = 8 waves in 4(M) x 2(N); WG tile 128x128; K step 32.
//    Both tiles async-copied to LDS (4 x b128 per thread per K-tile).
// ---------------------------------------------------------------------------
__global__ __launch_bounds__(256) void qkv_gemm(
    const _Float16* __restrict__ X,   // [16384][1024] f16, row-major
    const _Float16* __restrict__ Wt,  // [1024 n][1024 k] f16 (pre-transposed)
    const float* __restrict__ bias,   // [1024] f32
    _Float16* __restrict__ Out,       // [64][4096][64] f16 head-split
    float scale) {
  __shared__ _Float16 sX[2][128][40];
  __shared__ _Float16 sW[2][128][40];

  const int t = threadIdx.x;
  const int lane = t & 31;
  const int wid = t >> 5;
  const int waveM = wid >> 1;  // 0..3
  const int waveN = wid & 1;   // 0..1
  const int Mbase = blockIdx.y * 128;
  const int Nbase = blockIdx.x * 128;

  f8 acc[2][4];
#pragma unroll
  for (int mi = 0; mi < 2; mi++)
#pragma unroll
    for (int ni = 0; ni < 4; ni++) acc[mi][ni] = f8zero();

  auto issue_tile = [&](int kt, int buf) {
    const int k0 = kt * 32;
#pragma unroll
    for (int it = 0; it < 2; it++) {
      int slot = t + it * 256;  // 512 chunks of 8 f16 per operand
      int row = slot >> 2, ch = slot & 3;
      async_b128(&sX[buf][row][ch * 8],
                 X + (size_t)(Mbase + row) * DIMD + k0 + ch * 8);
      async_b128(&sW[buf][row][ch * 8],
                 Wt + (size_t)(Nbase + row) * DIMD + k0 + ch * 8);
    }
  };

  issue_tile(0, 0);

  const int r = lane & 15;
  const int hi = lane >> 4;

#pragma unroll 1
  for (int kt = 0; kt < 32; kt++) {
    const int buf = kt & 1;
    if (kt + 1 < 32) {
      issue_tile(kt + 1, buf ^ 1);
      // in-order completion: <=4 outstanding means tile kt has landed
      asm volatile("s_wait_asynccnt 0x4" ::: "memory");
    } else {
      asm volatile("s_wait_asynccnt 0x0" ::: "memory");
    }
    __syncthreads();  // all waves' current-tile data visible

    h16 Af[2], Bf[4];
#pragma unroll
    for (int mi = 0; mi < 2; mi++) {
      const int row = waveM * 32 + mi * 16 + r;
      Af[mi] = hcat(*(const h8*)&sX[buf][row][hi * 8],
                    *(const h8*)&sX[buf][row][16 + hi * 8]);
    }
#pragma unroll
    for (int ni = 0; ni < 4; ni++) {
      const int n = waveN * 64 + ni * 16 + r;
      const int ks = hi * 8;
      Bf[ni] = hcat(*(const h8*)&sW[buf][n][ks],
                    *(const h8*)&sW[buf][n][ks + 16]);
    }
#pragma unroll
    for (int mi = 0; mi < 2; mi++)
#pragma unroll
      for (int ni = 0; ni < 4; ni++)
        acc[mi][ni] = wmma_f16(Af[mi], Bf[ni], acc[mi][ni]);

    __syncthreads();  // buffer free before iteration kt+1 overwrites buf
  }

  // ---- epilogue: bias, scale, head-split f16 store ----
#pragma unroll
  for (int ni = 0; ni < 4; ni++) {
    const int col = Nbase + waveN * 64 + ni * 16 + r;
    const float bv = bias[col];
    const int h = col >> 6;
    const int d = col & 63;
#pragma unroll
    for (int mi = 0; mi < 2; mi++) {
#pragma unroll
      for (int v = 0; v < 8; v++) {
        const int grow = Mbase + waveM * 32 + mi * 16 + v + 8 * hi;
        const int bidx = grow >> 12;  // batch
        const int tok = grow & 4095;  // token
        const float val = (acc[mi][ni][v] + bv) * scale;
        Out[(((size_t)(bidx * HH + h)) * NN + tok) * DHD + d] = (_Float16)val;
      }
    }
  }
}

// ---------------------------------------------------------------------------
// 4) coarsen one level: q,k mean-pool by 2; v sum-pool by 2
// ---------------------------------------------------------------------------
__global__ void coarsen(const _Float16* __restrict__ qi,
                        const _Float16* __restrict__ ki,
                        const _Float16* __restrict__ vi,
                        _Float16* __restrict__ qo, _Float16* __restrict__ ko,
                        _Float16* __restrict__ vo, int nout) {
  size_t idx = (size_t)blockIdx.x * blockDim.x + threadIdx.x;
  size_t total = (size_t)BHD * nout * DHD;
  if (idx >= total) return;
  size_t d = idx & 63;
  size_t tk = (idx >> 6) % (size_t)nout;
  size_t bh = idx / ((size_t)nout * DHD);
  size_t i0 = (bh * (size_t)(nout * 2) + 2 * tk) * DHD + d;
  size_t i1 = i0 + DHD;
  qo[idx] = (_Float16)(((float)qi[i0] + (float)qi[i1]) * 0.5f);
  ko[idx] = (_Float16)(((float)ki[i0] + (float)ki[i1]) * 0.5f);
  vo[idx] = (_Float16)((float)vi[i0] + (float)vi[i1]);
}

// ---------------------------------------------------------------------------
// 5) one attention level: 1 wave per 16-token block, 4 waves / WG.
// ---------------------------------------------------------------------------
__global__ __launch_bounds__(128) void hatt_level(
    const _Float16* __restrict__ q,  // [64][nl][64]
    const _Float16* __restrict__ k,
    const _Float16* __restrict__ v,
    const float* __restrict__ Yprev,  // [64][nl/2][64] or null (coarsest)
    const float* __restrict__ Aprev,  // [64][nl/2]     or null
    float* __restrict__ Yout,         // [64][nl][64]
    float* __restrict__ Aout,         // [64][nl]
    int nl, int flip) {
  __shared__ _Float16 expA[4][16][16];
  __shared__ _Float16 vst[4][16][64];

  const int lane = threadIdx.x & 31;
  const int w = threadIdx.x >> 5;
  const int wave_global = blockIdx.x * 4 + w;
  const int blocks_per_bh = nl >> 4;
  const int bh = wave_global / blocks_per_bh;
  const int blk = wave_global % blocks_per_bh;
  const int kvblk = flip ? (blk ^ 1) : blk;

  const size_t qbase = ((size_t)bh * nl + (size_t)blk * 16) * DHD;
  const size_t kvbase = ((size_t)bh * nl + (size_t)kvblk * 16) * DHD;

  const int r = lane & 15;
  const int hi = lane >> 4;

  // ---- S = q k^T : two 16x16x32 WMMAs over K = 64 ----
  f8 S = f8zero();
#pragma unroll
  for (int kt = 0; kt < 2; kt++) {
    h16 Aq = hcat(*(const h8*)(q + qbase + (size_t)r * DHD + kt * 32 + hi * 8),
                  *(const h8*)(q + qbase + (size_t)r * DHD + kt * 32 + 16 + hi * 8));
    h16 Bk = hcat(*(const h8*)(k + kvbase + (size_t)r * DHD + kt * 32 + hi * 8),
                  *(const h8*)(k + kvbase + (size_t)r * DHD + kt * 32 + 16 + hi * 8));
    S = wmma_f16(Aq, Bk, S);
  }

  // ---- stage v block (16x64 f16) into LDS row-major ----
#pragma unroll
  for (int it = 0; it < 4; it++) {
    int chunk = lane + it * 32;  // 128 chunks of 8 f16
    int row = chunk >> 3, off = (chunk & 7) * 8;
    *(h8*)&vst[w][row][off] = *(const h8*)(v + kvbase + (size_t)row * DHD + off);
  }

  // ---- rowmax, exp, rowsum (C-layout: row i+8*hi, cols across half-wave) ----
  float rowsum[8];
#pragma unroll
  for (int i = 0; i < 8; i++) {
    float m = S[i];
    m = fmaxf(m, __shfl_xor(m, 1, 32));
    m = fmaxf(m, __shfl_xor(m, 2, 32));
    m = fmaxf(m, __shfl_xor(m, 4, 32));
    m = fmaxf(m, __shfl_xor(m, 8, 32));
    float e = __expf(S[i] - m);
    expA[w][i + 8 * hi][r] = (_Float16)e;
    float s = e;
    s += __shfl_xor(s, 1, 32);
    s += __shfl_xor(s, 2, 32);
    s += __shfl_xor(s, 4, 32);
    s += __shfl_xor(s, 8, 32);
    rowsum[i] = s;
  }
  __syncthreads();

  // ---- A' fragment: 16x32 f16, K = 0..15 valid, 16..31 zero ----
  h16 Af;
#pragma unroll
  for (int e = 0; e < 16; e++) Af[e] = hi ? (_Float16)0.0f : expA[w][r][e];

  // ---- y = A' v : 4 output tiles of 16 cols each ----
#pragma unroll
  for (int tT = 0; tT < 4; tT++) {
    h16 Bv;
    if (hi == 0) {  // lanes 0..15 hold K = 0..15 (columns of v)
#pragma unroll
      for (int e = 0; e < 16; e++) Bv[e] = vst[w][e][tT * 16 + r];
    } else {        // lanes 16..31 hold K = 16..31 -> zero padding
#pragma unroll
      for (int e = 0; e < 16; e++) Bv[e] = (_Float16)0.0f;
    }
    f8 y = wmma_f16(Af, Bv, f8zero());

#pragma unroll
    for (int i = 0; i < 8; i++) {
      const int row = i + 8 * hi;  // token within block
      const int tok = blk * 16 + row;
      const int d = tT * 16 + r;
      float val = y[i];
      if (Yprev)
        val += Yprev[((size_t)bh * (nl >> 1) + (tok >> 1)) * DHD + d];
      Yout[((size_t)bh * nl + tok) * DHD + d] = val;
    }
  }

  // ---- denominators ----
  if (r == 0) {  // lanes 0 and 16 write 8 rows each
#pragma unroll
    for (int i = 0; i < 8; i++) {
      const int tok = blk * 16 + i + 8 * hi;
      float s = rowsum[i];
      if (Aprev) s += Aprev[(size_t)bh * (nl >> 1) + (tok >> 1)];
      Aout[(size_t)bh * nl + tok] = s;
    }
  }
}

// ---------------------------------------------------------------------------
// 6) finalize: out[b][tok][h*64+d] = Y[bh][tok][d] / (A[bh][tok] + eps)
// ---------------------------------------------------------------------------
__global__ void finalize(const float* __restrict__ Y,
                         const float* __restrict__ A,
                         float* __restrict__ out) {
  size_t idx = (size_t)blockIdx.x * blockDim.x + threadIdx.x;
  const int d = (int)(idx & 63);
  const int h = (int)((idx >> 6) & 15);
  const size_t tok = (idx >> 10) & 4095;
  const size_t b = idx >> 22;
  const size_t bh = b * HH + h;
  const float y = Y[(bh * NN + tok) * DHD + d];
  const float a = A[bh * NN + tok];
  out[idx] = y / (a + 1e-8f);
}

// ---------------------------------------------------------------------------
// host-side orchestration
// ---------------------------------------------------------------------------
extern "C" void kernel_launch(void* const* d_in, const int* in_sizes, int n_in,
                              void* d_out, int out_size, void* d_ws,
                              size_t ws_size, hipStream_t stream) {
  const float* hs = (const float*)d_in[0];
  const float* Wq = (const float*)d_in[1];
  const float* bq = (const float*)d_in[2];
  const float* Wk = (const float*)d_in[3];
  const float* bk = (const float*)d_in[4];
  const float* Wv = (const float*)d_in[5];
  const float* bv = (const float*)d_in[6];

  char* ws = (char*)d_ws;
  size_t off = 0;
  auto alloc = [&](size_t bytes) -> void* {
    void* p = ws + off;
    off = (off + bytes + 255) & ~(size_t)255;
    return p;
  };

  // per-tensor level offsets (f16 elements), levels 0 (n=4096) .. 6 (n=64)
  size_t lvl_off[8];
  lvl_off[0] = 0;
  for (int l = 0; l < 7; l++)
    lvl_off[l + 1] = lvl_off[l] + (size_t)BHD * (size_t)(NN >> l) * DHD;
  const size_t qkv_elems = lvl_off[7];

  _Float16* Xh = (_Float16*)alloc((size_t)MROWS * DIMD * sizeof(_Float16));
  _Float16* Wtb = (_Float16*)alloc((size_t)DIMD * DIMD * sizeof(_Float16));
  _Float16* qlv = (_Float16*)alloc(qkv_elems * sizeof(_Float16));
  _Float16* klv = (_Float16*)alloc(qkv_elems * sizeof(_Float16));
  _Float16* vlv = (_Float16*)alloc(qkv_elems * sizeof(_Float16));
  float* Ybuf[2];
  float* Abuf[2];
  Ybuf[0] = (float*)alloc((size_t)BHD * NN * DHD * sizeof(float));
  Ybuf[1] = (float*)alloc((size_t)BHD * NN * DHD * sizeof(float));
  Abuf[0] = (float*)alloc((size_t)BHD * NN * sizeof(float));
  Abuf[1] = (float*)alloc((size_t)BHD * NN * sizeof(float));

  // 1) convert hidden states to f16
  {
    size_t n8 = (size_t)MROWS * DIMD / 8;  // 2,097,152
    cvt_f32_f16<<<(unsigned)(n8 / 256), 256, 0, stream>>>(hs, Xh, n8);
  }

  // 2/3) per projection: transpose+convert W, then WMMA GEMM
  {
    dim3 tgrid(DIMD / 64, DIMD / 64);    // (16, 16)
    dim3 ggrid(DIMD / 128, MROWS / 128); // (8, 128)

    transpose_w<<<tgrid, 256, 0, stream>>>(Wq, Wtb);
    qkv_gemm<<<ggrid, 256, 0, stream>>>(Xh, Wtb, bq, qlv, 0.125f);

    transpose_w<<<tgrid, 256, 0, stream>>>(Wk, Wtb);
    qkv_gemm<<<ggrid, 256, 0, stream>>>(Xh, Wtb, bk, klv, 1.0f);

    transpose_w<<<tgrid, 256, 0, stream>>>(Wv, Wtb);
    qkv_gemm<<<ggrid, 256, 0, stream>>>(Xh, Wtb, bv, vlv, 1.0f);
  }

  // 4) coarsen 6 times
  for (int l = 0; l < 6; l++) {
    int nout = NN >> (l + 1);
    size_t total = (size_t)BHD * nout * DHD;
    unsigned blocks = (unsigned)((total + 255) / 256);
    coarsen<<<blocks, 256, 0, stream>>>(qlv + lvl_off[l], klv + lvl_off[l],
                                        vlv + lvl_off[l], qlv + lvl_off[l + 1],
                                        klv + lvl_off[l + 1],
                                        vlv + lvl_off[l + 1], nout);
  }

  // 5) attention per level, coarsest (lv=6) to finest (lv=0), fused upsample
  int ping = 0;
  const float* Yp = nullptr;
  const float* Ap = nullptr;
  for (int lv = 6; lv >= 0; lv--) {
    int nl = NN >> lv;
    int waves = BHD * (nl >> 4);  // one wave per block; divisible by 4
    hatt_level<<<(unsigned)(waves / 4), 128, 0, stream>>>(
        qlv + lvl_off[lv], klv + lvl_off[lv], vlv + lvl_off[lv], Yp, Ap,
        Ybuf[ping], Abuf[ping], nl, (lv != 0) ? 1 : 0);
    Yp = Ybuf[ping];
    Ap = Abuf[ping];
    ping ^= 1;
  }

  // 6) normalize + head merge into d_out (f32, [4][4096][1024])
  {
    size_t total = (size_t)BB * NN * DIMD;  // 16,777,216
    finalize<<<(unsigned)(total / 256), 256, 0, stream>>>(Yp, Ap,
                                                          (float*)d_out);
  }
}
